// EnhancedGATv2_2147483648535
// MI455X (gfx1250) — compile-verified, matrix operand order checked by reading, not run on hardware
//
#include <hip/hip_runtime.h>
#include <hip/hip_bf16.h>
#include <cstdint>

// ---------------------------------------------------------------------------
// EnhancedGATv2 on gfx1250 (MI455X)
//   - WMMA bf16 GEMMs with pre-packed weight fragments (no cvt in hot loop)
//   - bf16-resident activations between GEMMs
//   - L2-resident edge phases with native u32-max / f32-add atomics
// ---------------------------------------------------------------------------

typedef __attribute__((ext_vector_type(16))) __bf16 v16bf;
typedef __attribute__((ext_vector_type(8)))  __bf16 v8bf;
typedef __attribute__((ext_vector_type(4)))  __bf16 v4bf;
typedef __attribute__((ext_vector_type(8)))  float  v8f;

#define NEG_SLOPE 0.2f

// ---- helpers ---------------------------------------------------------------

__device__ __forceinline__ unsigned fflip(float f) {
    unsigned u = __float_as_uint(f);
    return (u & 0x80000000u) ? ~u : (u | 0x80000000u);
}
__device__ __forceinline__ float funflip(unsigned u) {
    return __uint_as_float((u & 0x80000000u) ? (u ^ 0x80000000u) : ~u);
}
__device__ __forceinline__ void atomic_max_u32(unsigned* p, unsigned v) {
    __hip_atomic_fetch_max(p, v, __ATOMIC_RELAXED, __HIP_MEMORY_SCOPE_AGENT);
}
__device__ __forceinline__ void atomic_add_f32(float* p, float v) {
    __hip_atomic_fetch_add(p, v, __ATOMIC_RELAXED, __HIP_MEMORY_SCOPE_AGENT);
}
__device__ __forceinline__ float elu_f(float v) {
    return v > 0.f ? v : (__expf(v) - 1.f);
}
__device__ __forceinline__ float lrelu_f(float v) {
    return v > 0.f ? v : NEG_SLOPE * v;
}

// ---- weight pre-pack: fp32 [kin,kout] -> bf16 WMMA B-fragments -------------
// Fragment f = tc*(kin/32)+kc holds, for each lane, 16 bf16 values in the
// exact per-lane order the 16-bit B operand wants (lanes 0-15: K {0..7,16..23}
// of column n0+lane; lanes 16-31: K {8..15,24..31}).
__global__ __launch_bounds__(256)
void pack_weights(const float* __restrict__ W, __bf16* __restrict__ out,
                  int kin, int kout) {
    int idx = blockIdx.x * blockDim.x + threadIdx.x;
    if (idx >= kin * kout) return;
    const int KC = kin >> 5;
    int j    = idx & 15;
    int lane = (idx >> 4) & 31;
    int f    = idx >> 9;
    int kc   = f % KC;
    int tc   = f / KC;
    int lr   = lane & 15;
    int lh   = (lane & 16) ? 8 : 0;
    int k    = kc * 32 + ((j < 8) ? (lh + j) : (16 + lh + (j - 8)));
    int col  = tc * 16 + lr;
    out[idx] = (__bf16)W[(size_t)k * kout + col];
}

// ---- fp32 -> bf16 copy (4 elems/thread) ------------------------------------
__global__ __launch_bounds__(256)
void cvt_f32_bf16(const float* __restrict__ in, __bf16* __restrict__ out,
                  int total4) {
    int i = blockIdx.x * blockDim.x + threadIdx.x;
    if (i >= total4) return;
    float4 v = ((const float4*)in)[i];
    v4bf o;
    o[0] = (__bf16)v.x; o[1] = (__bf16)v.y; o[2] = (__bf16)v.z; o[3] = (__bf16)v.w;
    ((v4bf*)out)[i] = o;
}

// ---- WMMA GEMM: C[n,kout] = act(A_bf16[n,KIN] @ Wpack + bias) --------------
// One wave = one 16-col tile x 4 row tiles (64 rows); B fragment loaded once
// per K-chunk as a contiguous 32B lane read and reused by 4 WMMAs.
template<int KIN, bool BIAS, bool ELU, bool OUTBF16>
__global__ __launch_bounds__(256)
void gemm_wmma(const __bf16* __restrict__ A, const __bf16* __restrict__ Wp,
               const float* __restrict__ bias, void* __restrict__ Cout,
               int n, int kout) {
    const int wave = threadIdx.x >> 5;
    const int lane = threadIdx.x & 31;
    const int Tc   = kout >> 4;
    const int Trs  = (n + 63) >> 6;
    const int tile = blockIdx.x * 8 + wave;
    if (tile >= Trs * Tc) return;               // wave-uniform
    const int ts = tile / Tc;
    const int tc = tile % Tc;
    const int n0 = tc << 4;
    const int lr = lane & 15;
    const int lh = (lane & 16) ? 8 : 0;
    const int col = n0 + lr;
    const int KC  = KIN >> 5;

    const __bf16* arow[4];
    int m0[4];
#pragma unroll
    for (int t = 0; t < 4; ++t) {
        m0[t] = ((ts << 2) + t) << 4;
        int r = m0[t] + lr;
        if (r >= n) r = n - 1;                  // clamp loads in-bounds
        arow[t] = A + (size_t)r * KIN;
    }
    const __bf16* wfrag = Wp + ((size_t)tc * KC * 32 + lane) * 16;

    v8f acc[4] = {v8f{}, v8f{}, v8f{}, v8f{}};

#pragma unroll
    for (int kc = 0; kc < KC; ++kc) {
        const int kk = kc << 5;
        v16bf bm = *(const v16bf*)(wfrag + (size_t)kc * 32 * 16);
#pragma unroll
        for (int t = 0; t < 4; ++t) {
            v8bf alo = *(const v8bf*)(arow[t] + kk + lh);
            v8bf ahi = *(const v8bf*)(arow[t] + kk + 16 + lh);
            v16bf am = __builtin_shufflevector(alo, ahi,
                0, 1, 2, 3, 4, 5, 6, 7, 8, 9, 10, 11, 12, 13, 14, 15);
            acc[t] = __builtin_amdgcn_wmma_f32_16x16x32_bf16(
                false, am, false, bm, (short)0, acc[t], false, false);
        }
    }

    const float bv   = BIAS ? bias[col] : 0.f;
    const int   roff = (lane & 16) ? 8 : 0;
#pragma unroll
    for (int t = 0; t < 4; ++t) {
        if (m0[t] >= n) continue;               // wave-uniform
#pragma unroll
        for (int r = 0; r < 8; ++r) {
            int row = m0[t] + roff + r;
            if (row < n) {
                float v = acc[t][r] + bv;
                if (ELU) v = elu_f(v);
                if (OUTBF16)
                    ((__bf16*)Cout)[(size_t)row * kout + col] = (__bf16)v;
                else
                    ((float*)Cout)[(size_t)row * kout + col] = v;
            }
        }
    }
}

// ---- edge pass A: logits + segment max ------------------------------------
template<int H, int C>
__global__ __launch_bounds__(256)
void edge_logits(const float* __restrict__ xl, const float* __restrict__ xr,
                 const float* __restrict__ att, const int* __restrict__ ei,
                 int E, int Etot, float* __restrict__ logit,
                 unsigned* __restrict__ mmax) {
    int idx = blockIdx.x * blockDim.x + threadIdx.x;
    if (idx >= Etot * H) return;
    int e = idx / H, h = idx % H;
    int src, dst;
    if (e < E) { src = ei[e]; dst = ei[E + e]; } else { src = dst = e - E; }
    const float* pl = xl + (size_t)src * (H * C) + h * C;
    const float* pr = xr + (size_t)dst * (H * C) + h * C;
    const float* pa = att + h * C;
    float acc = 0.f;
#pragma unroll
    for (int c = 0; c < C; c += 4) {
        float4 a = *(const float4*)(pl + c);
        float4 b = *(const float4*)(pr + c);
        float4 w = *(const float4*)(pa + c);
        acc += w.x * lrelu_f(a.x + b.x);
        acc += w.y * lrelu_f(a.y + b.y);
        acc += w.z * lrelu_f(a.z + b.z);
        acc += w.w * lrelu_f(a.w + b.w);
    }
    logit[idx] = acc;
    atomic_max_u32(&mmax[(size_t)dst * H + h], fflip(acc));
}

// ---- edge pass B: p = exp(logit - max), segment sum -----------------------
template<int H>
__global__ __launch_bounds__(256)
void edge_expsum(const int* __restrict__ ei, int E, int Etot,
                 float* __restrict__ logit, const unsigned* __restrict__ mmax,
                 float* __restrict__ ssum) {
    int idx = blockIdx.x * blockDim.x + threadIdx.x;
    if (idx >= Etot * H) return;
    int e = idx / H, h = idx % H;
    int dst = (e < E) ? ei[E + e] : (e - E);
    float m = funflip(mmax[(size_t)dst * H + h]);
    float p = __expf(logit[idx] - m);
    logit[idx] = p;
    atomic_add_f32(&ssum[(size_t)dst * H + h], p);
}

// ---- edge pass C: agg[dst] += xl[src] * alpha -----------------------------
template<int H, int C>
__global__ __launch_bounds__(256)
void edge_aggregate(const float* __restrict__ xl, const int* __restrict__ ei,
                    int E, int Etot, const float* __restrict__ p,
                    const float* __restrict__ ssum, float* __restrict__ agg) {
    int idx = blockIdx.x * blockDim.x + threadIdx.x;
    const int HC_ = H * C;
    if (idx >= Etot * HC_) return;
    int e = idx / HC_, c = idx % HC_;
    int h = c / C;
    int src, dst;
    if (e < E) { src = ei[e]; dst = ei[E + e]; } else { src = dst = e - E; }
    float alpha = p[(size_t)e * H + h] / ssum[(size_t)dst * H + h];
    atomic_add_f32(&agg[(size_t)dst * HC_ + c],
                   xl[(size_t)src * HC_ + c] * alpha);
}

// ---- node-wise bias + ELU, fp32 in -> bf16 out -----------------------------
__global__ __launch_bounds__(256)
void bias_elu_bf16(const float* __restrict__ h, const float* __restrict__ b,
                   __bf16* __restrict__ out, int n, int k) {
    int idx = blockIdx.x * blockDim.x + threadIdx.x;
    if (idx >= n * k) return;
    out[idx] = (__bf16)elu_f(h[idx] + b[idx % k]);
}

// ---- final: bias + log_softmax over 16 channels ---------------------------
__global__ __launch_bounds__(256)
void finalize_logsoftmax(const float* __restrict__ agg,
                         const float* __restrict__ b2,
                         float* __restrict__ out, int n) {
    int i = blockIdx.x * blockDim.x + threadIdx.x;
    if (i >= n) return;
    float v[16];
    float m = -3.402823466e38f;
#pragma unroll
    for (int c = 0; c < 16; ++c) {
        v[c] = agg[(size_t)i * 16 + c] + b2[c];
        m = fmaxf(m, v[c]);
    }
    float s = 0.f;
#pragma unroll
    for (int c = 0; c < 16; ++c) s += __expf(v[c] - m);
    float l = __logf(s);
#pragma unroll
    for (int c = 0; c < 16; ++c) out[(size_t)i * 16 + c] = v[c] - m - l;
}

// ---------------------------------------------------------------------------

extern "C" void kernel_launch(void* const* d_in, const int* in_sizes, int n_in,
                              void* d_out, int out_size, void* d_ws, size_t ws_size,
                              hipStream_t stream) {
    const float* x    = (const float*)d_in[0];
    const int*   ei   = (const int*)d_in[1];
    const float* Wl1  = (const float*)d_in[2];
    const float* Wr1  = (const float*)d_in[3];
    const float* att1 = (const float*)d_in[4];
    const float* b1   = (const float*)d_in[5];
    const float* Wk   = (const float*)d_in[6];
    const float* bk   = (const float*)d_in[7];
    const float* Wl2  = (const float*)d_in[8];
    const float* Wr2  = (const float*)d_in[9];
    const float* att2 = (const float*)d_in[10];
    const float* b2   = (const float*)d_in[11];
    float* out = (float*)d_out;

    const int N    = in_sizes[0] / 128;   // 50000
    const int E    = in_sizes[1] / 2;     // 800000
    const int Etot = E + N;               // self-loops appended

    // ---- workspace layout ----
    // fp32 region
    float*    xl1   = (float*)d_ws;                          // N*128
    float*    xr1   = xl1 + (size_t)N * 128;                 // N*128 (later hl2/hr2)
    float*    hbuf  = xr1 + (size_t)N * 128;                 // N*128 (agg1)
    float*    logit = hbuf + (size_t)N * 128;                // Etot*8 (reused conv2)
    unsigned* mmax  = (unsigned*)(logit + (size_t)Etot * 8); // N*8
    float*    ssum  = (float*)(mmax + (size_t)N * 8);        // N*8
    float*    agg2  = ssum + (size_t)N * 8;                  // N*16
    // bf16 region (16B-aligned: fp32 region is a multiple of 16 bytes)
    __bf16*   xbf   = (__bf16*)(agg2 + (size_t)N * 16);      // N*128 (also hbf0)
    __bf16*   hbf0  = xbf;
    __bf16*   hbf1  = xbf + (size_t)N * 128;                 // N*128
    __bf16*   wpack = hbf1 + (size_t)N * 128;                // packed weights
    __bf16*   Wl1p  = wpack;                                 // 128*128
    __bf16*   Wr1p  = Wl1p + 128 * 128;
    __bf16*   Wk0p  = Wr1p + 128 * 128;
    __bf16*   Wk1p  = Wk0p + 128 * 128;
    __bf16*   Wk2p  = Wk1p + 128 * 128;
    __bf16*   Wl2p  = Wk2p + 128 * 128;                      // 128*16
    __bf16*   Wr2p  = Wl2p + 128 * 16;

    float*    hl2 = xr1;                                     // N*16
    float*    hr2 = xr1 + (size_t)N * 16;                    // N*16

    const int threads = 256;
    auto cdiv = [](int a, int b) { return (a + b - 1) / b; };
    auto gemm_blocks = [&](int n, int kout) {
        int tiles = ((n + 63) >> 6) * (kout >> 4);
        return (tiles + 7) / 8;
    };

    // ---- pre-pack weights (tiny) + convert x to bf16 ----
    hipLaunchKernelGGL(pack_weights, dim3(cdiv(128 * 128, 256)), dim3(threads),
                       0, stream, Wl1, Wl1p, 128, 128);
    hipLaunchKernelGGL(pack_weights, dim3(cdiv(128 * 128, 256)), dim3(threads),
                       0, stream, Wr1, Wr1p, 128, 128);
    hipLaunchKernelGGL(pack_weights, dim3(cdiv(128 * 128, 256)), dim3(threads),
                       0, stream, Wk, Wk0p, 128, 128);
    hipLaunchKernelGGL(pack_weights, dim3(cdiv(128 * 128, 256)), dim3(threads),
                       0, stream, Wk + 128 * 128, Wk1p, 128, 128);
    hipLaunchKernelGGL(pack_weights, dim3(cdiv(128 * 128, 256)), dim3(threads),
                       0, stream, Wk + 2 * 128 * 128, Wk2p, 128, 128);
    hipLaunchKernelGGL(pack_weights, dim3(cdiv(128 * 16, 256)), dim3(threads),
                       0, stream, Wl2, Wl2p, 128, 16);
    hipLaunchKernelGGL(pack_weights, dim3(cdiv(128 * 16, 256)), dim3(threads),
                       0, stream, Wr2, Wr2p, 128, 16);
    hipLaunchKernelGGL(cvt_f32_bf16, dim3(cdiv(N * 32, 256)), dim3(threads),
                       0, stream, x, xbf, N * 32);

    // ---- conv1 transforms: xl1 = x@Wl1, xr1 = x@Wr1 (fp32 out) ----
    hipLaunchKernelGGL((gemm_wmma<128, false, false, false>),
                       dim3(gemm_blocks(N, 128)), dim3(threads), 0, stream,
                       xbf, Wl1p, nullptr, xl1, N, 128);
    hipLaunchKernelGGL((gemm_wmma<128, false, false, false>),
                       dim3(gemm_blocks(N, 128)), dim3(threads), 0, stream,
                       xbf, Wr1p, nullptr, xr1, N, 128);

    // ---- conv1 attention (segment softmax + scatter) ----
    hipMemsetAsync(mmax, 0, (size_t)N * 8 * sizeof(unsigned), stream);
    hipMemsetAsync(ssum, 0, (size_t)N * 8 * sizeof(float), stream);
    hipMemsetAsync(hbuf, 0, (size_t)N * 128 * sizeof(float), stream);

    int w1 = Etot * 8;
    hipLaunchKernelGGL((edge_logits<8, 16>), dim3(cdiv(w1, 256)), dim3(threads),
                       0, stream, xl1, xr1, att1, ei, E, Etot, logit, mmax);
    hipLaunchKernelGGL((edge_expsum<8>), dim3(cdiv(w1, 256)), dim3(threads),
                       0, stream, ei, E, Etot, logit, mmax, ssum);
    int w1c = Etot * 128;
    hipLaunchKernelGGL((edge_aggregate<8, 16>), dim3(cdiv(w1c, 256)), dim3(threads),
                       0, stream, xl1, ei, E, Etot, logit, ssum, hbuf);
    // hbuf = conv1 aggregate; h = elu(hbuf + b1) stored as bf16 (overwrites xbf)
    hipLaunchKernelGGL(bias_elu_bf16, dim3(cdiv(N * 128, 256)), dim3(threads),
                       0, stream, hbuf, b1, hbf0, N, 128);

    // ---- knowledge layers: h = elu(h @ Wk[i] + bk[i]), bf16 ping-pong ----
    hipLaunchKernelGGL((gemm_wmma<128, true, true, true>),
                       dim3(gemm_blocks(N, 128)), dim3(threads), 0, stream,
                       hbf0, Wk0p, bk, (void*)hbf1, N, 128);
    hipLaunchKernelGGL((gemm_wmma<128, true, true, true>),
                       dim3(gemm_blocks(N, 128)), dim3(threads), 0, stream,
                       hbf1, Wk1p, bk + 128, (void*)hbf0, N, 128);
    hipLaunchKernelGGL((gemm_wmma<128, true, true, true>),
                       dim3(gemm_blocks(N, 128)), dim3(threads), 0, stream,
                       hbf0, Wk2p, bk + 256, (void*)hbf1, N, 128);
    // final hidden state (bf16) in hbf1

    // ---- conv2 transforms: hl2 = h@Wl2, hr2 = h@Wr2 (fp32 out) ----
    hipLaunchKernelGGL((gemm_wmma<128, false, false, false>),
                       dim3(gemm_blocks(N, 16)), dim3(threads), 0, stream,
                       hbf1, Wl2p, nullptr, hl2, N, 16);
    hipLaunchKernelGGL((gemm_wmma<128, false, false, false>),
                       dim3(gemm_blocks(N, 16)), dim3(threads), 0, stream,
                       hbf1, Wr2p, nullptr, hr2, N, 16);

    // ---- conv2 attention ----
    hipMemsetAsync(mmax, 0, (size_t)N * sizeof(unsigned), stream);
    hipMemsetAsync(ssum, 0, (size_t)N * sizeof(float), stream);
    hipMemsetAsync(agg2, 0, (size_t)N * 16 * sizeof(float), stream);

    int w2 = Etot;
    hipLaunchKernelGGL((edge_logits<1, 16>), dim3(cdiv(w2, 256)), dim3(threads),
                       0, stream, hl2, hr2, att2, ei, E, Etot, logit, mmax);
    hipLaunchKernelGGL((edge_expsum<1>), dim3(cdiv(w2, 256)), dim3(threads),
                       0, stream, ei, E, Etot, logit, mmax, ssum);
    int w2c = Etot * 16;
    hipLaunchKernelGGL((edge_aggregate<1, 16>), dim3(cdiv(w2c, 256)), dim3(threads),
                       0, stream, hl2, ei, E, Etot, logit, ssum, agg2);

    // ---- bias + log_softmax ----
    hipLaunchKernelGGL(finalize_logsoftmax, dim3(cdiv(N, 256)), dim3(threads),
                       0, stream, agg2, b2, out, N);
}
